// CNN_48799418417833
// MI455X (gfx1250) — matrix operand order repaired and checked by reading, not compiled, not run on hardware
//
#include <hip/hip_runtime.h>
#include <hip/hip_bf16.h>

typedef __attribute__((ext_vector_type(16))) __bf16 v16bf;
typedef __attribute__((ext_vector_type(8)))  float  v8f;
typedef __attribute__((ext_vector_type(4)))  unsigned tdm_u4;
typedef __attribute__((ext_vector_type(8)))  int      tdm_i8;
typedef __attribute__((ext_vector_type(4)))  int      tdm_i4;

// ---------------------------------------------------------------------------
// WMMA helpers (gfx1250, wave32). D = A(16x32 bf16) x B(32x16 bf16) + C(f32)
// ---------------------------------------------------------------------------
__device__ __forceinline__ v8f wmma_bf16(v16bf a, v16bf b, v8f c) {
  return __builtin_amdgcn_wmma_f32_16x16x32_bf16(
      /*neg_a=*/false, a, /*neg_b=*/false, b,
      /*c_mod=*/(short)0, c, /*reuse_a=*/false, /*reuse_b=*/false);
}

// A fragment: lane holds row M=lane&15. VGPR v (=j>>1) half (j&1):
//   k = (lane>=16 ? 16 : 0) + v*2 + (j&1)
// rowptr must point at a K-contiguous bf16 row; pairs coalesce to b128 loads.
__device__ __forceinline__ v16bf load_a_frag(const __bf16* rowptr, int hi8) {
  union { v16bf v; unsigned u[8]; } r;
#pragma unroll
  for (int t = 0; t < 8; ++t) {
    int k = (hi8 << 1) + (t << 1);
    r.u[t] = *(const unsigned*)(rowptr + k);
  }
  return r.v;
}

// B fragment: lane holds col N=lane&15, 16 contiguous K at +hi16 (0 or 16).
__device__ __forceinline__ v16bf load_b_frag(const __bf16* rowptr, int hi16) {
  return *(const v16bf*)(rowptr + hi16);
}

// ---------------------------------------------------------------------------
// TDM: stage a 2D tile (tile_rows x row_elems bf16, row-major, stride =
// row_elems) from global into LDS at lds_off. D# per CDNA5 ISA §8.
// This toolchain exposes the 6-arg builtin (extra zero descriptor group).
// ---------------------------------------------------------------------------
__device__ __forceinline__ void tdm_load_2d_to_lds(unsigned lds_off,
                                                   const void* gptr,
                                                   unsigned tensor_rows,
                                                   unsigned row_elems,
                                                   unsigned tile_rows) {
  unsigned long long ga = (unsigned long long)gptr;
  tdm_u4 g0;
  g0[0] = 1u;                                          // count=1, user D#
  g0[1] = lds_off;                                     // lds_addr (bytes)
  g0[2] = (unsigned)(ga & 0xffffffffu);                // global_addr[31:0]
  g0[3] = (unsigned)((ga >> 32) & 0x1ffffffu) | (2u << 30);  // [56:32] | type=2
  tdm_i8 g1;
  g1[0] = (int)(1u << 16);                             // wg_mask=0, data_size=2B
  g1[1] = (int)(row_elems << 16);                      // tensor_dim0[15:0]
  g1[2] = (int)((row_elems >> 16) | (tensor_rows << 16));   // dim0 hi | dim1 lo
  g1[3] = (int)((tensor_rows >> 16) | (row_elems << 16));   // dim1 hi | tile_dim0
  g1[4] = (int)tile_rows;                              // tile_dim1, tile_dim2=0
  g1[5] = (int)row_elems;                              // tensor_dim0_stride lo
  g1[6] = 0;                                           // stride0 hi | stride1 lo
  g1[7] = 0;
  tdm_i4 z4 = {0, 0, 0, 0};                            // 2D tile: groups 2/3 unused
  tdm_i8 z8 = {0, 0, 0, 0, 0, 0, 0, 0};
  __builtin_amdgcn_tensor_load_to_lds(g0, g1, z4, z4, z8, 0);
}

#define Bn 8
#define Ln 2048
#define Hn 128
#define BLn (Bn * Ln)
#define DINn (3 * Hn)

// ---------------------------------------------------------------------------
// Kernel 0: transpose weights to bf16, K-contiguous per output channel.
// ---------------------------------------------------------------------------
__global__ void k_prep_w(const float* __restrict__ w1, const float* __restrict__ w2,
                         const float* __restrict__ cw,
                         __bf16* __restrict__ w1T, __bf16* __restrict__ w2T,
                         __bf16* __restrict__ cwT) {
  int idx = blockIdx.x * blockDim.x + threadIdx.x;
  if (idx >= DINn * Hn) return;
  int h = idx / DINn, k = idx % DINn;                  // wT[h][k] = w[k][h]
  w1T[idx] = (__bf16)w1[k * Hn + h];
  w2T[idx] = (__bf16)w2[k * Hn + h];
  int tap = idx / (Hn * Hn), r = idx % (Hn * Hn);
  int o = r / Hn, i = r % Hn;                          // cwT[tap][o][i] = cw[tap][i][o]
  cwT[idx] = (__bf16)cw[tap * Hn * Hn + i * Hn + o];
}

// ---------------------------------------------------------------------------
// Kernel 1: Shat bf16; x -> xT bf16 [B][H][L]; hcat thirds 0 and 2.
// ---------------------------------------------------------------------------
__global__ void __launch_bounds__(256) k_prep_act(
    const float* __restrict__ x, const float* __restrict__ sk,
    const float* __restrict__ ind, __bf16* __restrict__ sbf,
    __bf16* __restrict__ xT, __bf16* __restrict__ hcat) {
  int wid = threadIdx.x >> 5, lane = threadIdx.x & 31;
  int row = blockIdx.x * 8 + wid;
  int b = row / Ln, l = row % Ln;
  float s[4], xv[4], iv[4];
  float ss = 0.f;
#pragma unroll
  for (int i = 0; i < 4; ++i) {
    int h = lane + 32 * i;
    s[i]  = sk[(size_t)row * Hn + h];
    xv[i] = x[(size_t)row * Hn + h];
    iv[i] = ind[(size_t)row * Hn + h];
    ss += s[i] * s[i];
  }
#pragma unroll
  for (int off = 16; off > 0; off >>= 1) ss += __shfl_xor(ss, off);
  float rn = rsqrtf(ss);
#pragma unroll
  for (int i = 0; i < 4; ++i) {
    int h = lane + 32 * i;
    sbf[(size_t)row * Hn + h] = (__bf16)(s[i] * rn);
    xT[((size_t)b * Hn + h) * Ln + l] = (__bf16)xv[i];
    hcat[(size_t)row * DINn + h] = (__bf16)xv[i];
    hcat[(size_t)row * DINn + 2 * Hn + h] = (__bf16)iv[i];
  }
}

__device__ __forceinline__ bool att_valid(int l, int m) {
  // tril(k=2) minus eye (except (0,0)): valid iff m<=l+2 and (l!=m or l==0)
  return (m <= l + 2) && ((l != m) || (l == 0));
}

// ---------------------------------------------------------------------------
// Kernel 2: column sums D[b,m] = sum_l exp(cos[l,m]) over valid l.
// Fast path: tiles with lt > mt are fully inside the band (no mask VALU).
// ---------------------------------------------------------------------------
__global__ void __launch_bounds__(128) k_colsum(const __bf16* __restrict__ sbf,
                                                float* __restrict__ D) {
  int mt = blockIdx.x % (Ln / 16), b = blockIdx.x / (Ln / 16);
  int m0 = mt * 16;
  const __bf16* S = sbf + (size_t)b * Ln * Hn;
  int wid = threadIdx.x >> 5, lane = threadIdx.x & 31;
  int n = lane & 15, hi8 = (lane >> 4) << 3, hi16 = (lane >> 4) << 4;

  __shared__ float cs[16];
  if (threadIdx.x < 16) cs[threadIdx.x] = 0.f;
  __syncthreads();

  v16bf bf[4];                                         // B = Shat rows of m-tile
  const __bf16* brow = S + (size_t)(m0 + n) * Hn;
#pragma unroll
  for (int kc = 0; kc < 4; ++kc) bf[kc] = load_b_frag(brow + kc * 32, hi16);

  float part = 0.f;
  int m = m0 + n;
  int lt0 = (mt > 0) ? (mt - 1) : 0;                   // need l >= m0-2
  for (int lt = lt0 + wid; lt < Ln / 16; lt += 4) {
    int l0 = lt * 16;
    const __bf16* arow = S + (size_t)(l0 + n) * Hn;
    v8f c = {};
#pragma unroll
    for (int kc = 0; kc < 4; ++kc)
      c = wmma_bf16(load_a_frag(arow + kc * 32, hi8), bf[kc], c);
    if (lt > mt) {                                     // fully valid tile
#pragma unroll
      for (int r = 0; r < 8; ++r) part += __expf(c[r]);
    } else {                                           // band boundary tiles
#pragma unroll
      for (int r = 0; r < 8; ++r) {
        int l = l0 + r + hi8;
        if (att_valid(l, m)) part += __expf(c[r]);
      }
    }
  }
  atomicAdd(&cs[n], part);                             // ds_add_f32
  __syncthreads();
  if (threadIdx.x < 16) D[(size_t)b * Ln + m0 + threadIdx.x] = cs[threadIdx.x];
}

// ---------------------------------------------------------------------------
// Kernel 3: x2 = P @ x, P = exp(cos)*valid/D[m], fused in LDS (no HBM att).
// ---------------------------------------------------------------------------
__global__ void __launch_bounds__(128) k_attxv(const __bf16* __restrict__ sbf,
                                               const __bf16* __restrict__ xT,
                                               const float* __restrict__ D,
                                               __bf16* __restrict__ hcat) {
  int lt = blockIdx.x % (Ln / 16), b = blockIdx.x / (Ln / 16);
  int l0 = lt * 16;
  const __bf16* S = sbf + (size_t)b * Ln * Hn;
  const __bf16* XT = xT + (size_t)b * Hn * Ln;
  const float* Db = D + (size_t)b * Ln;
  int wid = threadIdx.x >> 5, lane = threadIdx.x & 31;
  int n = lane & 15, hi8 = (lane >> 4) << 3, hi16 = (lane >> 4) << 4;

  __shared__ __bf16 pb[2][16][32];                     // P tile, double buffered

  v8f acc0 = {}, acc1 = {};
  v16bf sa[4];                                         // Shat A-frags (wave0 only)
  if (wid == 0) {
    const __bf16* arow = S + (size_t)(l0 + n) * Hn;
#pragma unroll
    for (int kc = 0; kc < 4; ++kc) sa[kc] = load_a_frag(arow + kc * 32, hi8);
  }
  int h0 = wid * 32;
  const __bf16* xrow0 = XT + (size_t)(h0 + n) * Ln;
  const __bf16* xrow1 = XT + (size_t)(h0 + 16 + n) * Ln;

  int mlim = min(Ln, l0 + 18);                         // columns with any valid row
  int nch = (mlim + 31) >> 5;
  for (int ch = 0; ch < nch; ++ch) {
    int m0 = ch * 32;
    if (wid == 0) {
#pragma unroll
      for (int t = 0; t < 2; ++t) {
        int mb = m0 + t * 16;
        const __bf16* brow = S + (size_t)min(mb + n, Ln - 1) * Hn;
        v8f c = {};
#pragma unroll
        for (int kc = 0; kc < 4; ++kc)
          c = wmma_bf16(sa[kc], load_b_frag(brow + kc * 32, hi16), c);
        int m = mb + n;
        float rD = 1.f / Db[min(m, Ln - 1)];
        if (mb + 15 < l0) {                            // strictly-below-band tile
#pragma unroll
          for (int r = 0; r < 8; ++r)
            pb[ch & 1][r + hi8][t * 16 + n] = (__bf16)(__expf(c[r]) * rD);
        } else {                                       // boundary: mask per element
#pragma unroll
          for (int r = 0; r < 8; ++r) {
            int l = l0 + r + hi8;
            float p = (m < Ln && att_valid(l, m)) ? __expf(c[r]) * rD : 0.f;
            pb[ch & 1][r + hi8][t * 16 + n] = (__bf16)p;
          }
        }
      }
    }
    __builtin_prefetch(xrow0 + m0 + 32, 0, 1);         // global_prefetch_b8
    __syncthreads();
    v16bf pf = load_a_frag(&pb[ch & 1][n][0], hi8);    // lane row = l-local
    acc0 = wmma_bf16(pf, load_b_frag(xrow0 + m0, hi16), acc0);
    acc1 = wmma_bf16(pf, load_b_frag(xrow1 + m0, hi16), acc1);
  }
#pragma unroll
  for (int r = 0; r < 8; ++r) {
    int l = l0 + r + hi8;
    __bf16* dst = hcat + ((size_t)b * Ln + l) * DINn + Hn;   // middle third
    dst[h0 + n] = (__bf16)acc0[r];
    dst[h0 + 16 + n] = (__bf16)acc1[r];
  }
}

// ---------------------------------------------------------------------------
// Kernel 4: GLU. [BL,384]x[384,128] twice; sigmoid-gate; bf16 out for convs.
// ---------------------------------------------------------------------------
__global__ void __launch_bounds__(256) k_glu(const __bf16* __restrict__ hcat,
                                             const __bf16* __restrict__ w1T,
                                             const __bf16* __restrict__ w2T,
                                             const float* __restrict__ b1,
                                             const float* __restrict__ b2,
                                             __bf16* __restrict__ hout) {
  int wid = threadIdx.x >> 5, lane = threadIdx.x & 31;
  int row0 = blockIdx.x * 16, h0 = wid * 16;
  int n = lane & 15, hi8 = (lane >> 4) << 3, hi16 = (lane >> 4) << 4;
  v8f a1 = {}, a2 = {};
  const __bf16* arow = hcat + (size_t)(row0 + n) * DINn;
  const __bf16* w1r = w1T + (size_t)(h0 + n) * DINn;
  const __bf16* w2r = w2T + (size_t)(h0 + n) * DINn;
#pragma unroll
  for (int kc = 0; kc < 12; ++kc) {
    v16bf a = load_a_frag(arow + kc * 32, hi8);
    a1 = wmma_bf16(a, load_b_frag(w1r + kc * 32, hi16), a1);
    a2 = wmma_bf16(a, load_b_frag(w2r + kc * 32, hi16), a2);
  }
  int h = h0 + n;
  float bb1 = b1[h], bb2 = b2[h];
#pragma unroll
  for (int r = 0; r < 8; ++r) {
    float g = 1.f / (1.f + __expf(-(a1[r] + bb1)));
    float v = (a2[r] + bb2) * g;
    hout[(size_t)(row0 + r + hi8) * Hn + h] = (__bf16)v;
  }
}

// ---------------------------------------------------------------------------
// Kernel 5: causal conv K=3 (+ReLU). The 18x128 bf16 input row window for the
// whole block is staged once by the Tensor Data Mover (wave0 issues
// tensor_load_to_lds, waits TENSORcnt, barrier); the 3 taps then read
// overlapping A fragments from LDS. Batch-boundary rows are zeroed per lane.
// Final pass multiplies by next_skill and writes f32 to d_out.
// ---------------------------------------------------------------------------
__global__ void __launch_bounds__(256) k_conv(const __bf16* __restrict__ hin,
                                              const __bf16* __restrict__ cwT,
                                              const float* __restrict__ cb,
                                              __bf16* __restrict__ houtb,
                                              float* __restrict__ outf,
                                              const float* __restrict__ nsk,
                                              int final_) {
  int wid = threadIdx.x >> 5, lane = threadIdx.x & 31;
  int row0 = blockIdx.x * 16, h0 = wid * 16;
  int n = lane & 15, hi8 = (lane >> 4) << 3, hi16 = (lane >> 4) << 4;
  int l0 = row0 % Ln;

  __shared__ __attribute__((aligned(128))) __bf16 stile[18 * Hn];

  int off = (row0 >= 2) ? 2 : 0;                       // only global row 0 clips
  if (wid == 0) {
    tdm_load_2d_to_lds(0u, hin + (size_t)(row0 - off) * Hn,
                       (unsigned)BLn, (unsigned)Hn, (unsigned)(16 + off));
    __builtin_amdgcn_s_wait_tensorcnt(0);
  }
  __syncthreads();

  v8f acc = {};
#pragma unroll
  for (int tap = 0; tap < 3; ++tap) {
    int lsh = l0 + n - 2 + tap;                        // in-batch source row
    bool ok = (lsh >= 0);
    int lrow = n - 2 + tap + off;                      // LDS tile row
    const __bf16* arow = stile + (size_t)max(lrow, 0) * Hn;
    const __bf16* wr = cwT + ((size_t)tap * Hn + h0 + n) * Hn;
#pragma unroll
    for (int kc = 0; kc < 4; ++kc) {
      v16bf a = load_a_frag(arow + kc * 32, hi8);
      if (!ok) {
#pragma unroll
        for (int i = 0; i < 16; ++i) a[i] = (__bf16)0.f;   // causal zero pad
      }
      acc = wmma_bf16(a, load_b_frag(wr + kc * 32, hi16), acc);
    }
  }
  int h = h0 + n;
  float bb = cb[h];
#pragma unroll
  for (int r = 0; r < 8; ++r) {
    float v = fmaxf(acc[r] + bb, 0.f);
    size_t idx = (size_t)(row0 + r + hi8) * Hn + h;
    if (final_) outf[idx] = v * nsk[idx];
    else        houtb[idx] = (__bf16)v;
  }
}

// ---------------------------------------------------------------------------
extern "C" void kernel_launch(void* const* d_in, const int* in_sizes, int n_in,
                              void* d_out, int out_size, void* d_ws, size_t ws_size,
                              hipStream_t stream) {
  const float* x    = (const float*)d_in[0];
  const float* sk   = (const float*)d_in[1];
  const float* ind  = (const float*)d_in[2];
  const float* nsk  = (const float*)d_in[3];
  const float* gw1  = (const float*)d_in[4];
  const float* gb1  = (const float*)d_in[5];
  const float* gw2  = (const float*)d_in[6];
  const float* gb2  = (const float*)d_in[7];
  const float* cw   = (const float*)d_in[8];
  const float* cb   = (const float*)d_in[9];
  float* out = (float*)d_out;

  char* w = (char*)d_ws;
  size_t o = 0;
  auto alloc = [&](size_t bytes) {
    size_t r = o; o = (o + bytes + 255) & ~(size_t)255; return r;
  };
  __bf16* sbf  = (__bf16*)(w + alloc((size_t)BLn * Hn * 2));
  __bf16* xT   = (__bf16*)(w + alloc((size_t)BLn * Hn * 2 + 256)); // +B-frag tail slack
  __bf16* hcat = (__bf16*)(w + alloc((size_t)BLn * DINn * 2));
  float*  Dsum = (float*) (w + alloc((size_t)BLn * 4));
  __bf16* w1T  = (__bf16*)(w + alloc((size_t)DINn * Hn * 2));
  __bf16* w2T  = (__bf16*)(w + alloc((size_t)DINn * Hn * 2));
  __bf16* cwT  = (__bf16*)(w + alloc((size_t)3 * Hn * Hn * 2));
  __bf16* hb0  = (__bf16*)(w + alloc((size_t)BLn * Hn * 2));
  __bf16* hb1  = (__bf16*)(w + alloc((size_t)BLn * Hn * 2));

  k_prep_w<<<(DINn * Hn + 255) / 256, 256, 0, stream>>>(gw1, gw2, cw, w1T, w2T, cwT);
  k_prep_act<<<BLn / 8, 256, 0, stream>>>(x, sk, ind, sbf, xT, hcat);
  k_colsum<<<Bn * (Ln / 16), 128, 0, stream>>>(sbf, Dsum);
  k_attxv<<<Bn * (Ln / 16), 128, 0, stream>>>(sbf, xT, Dsum, hcat);
  k_glu<<<BLn / 16, 256, 0, stream>>>(hcat, w1T, w2T, gb1, gb2, hb0);
  k_conv<<<BLn / 16, 256, 0, stream>>>(hb0, cwT, cb, hb1, out, nsk, 0);
  k_conv<<<BLn / 16, 256, 0, stream>>>(hb1, cwT, cb, hb0, out, nsk, 0);
  k_conv<<<BLn / 16, 256, 0, stream>>>(hb0, cwT, cb, hb1, out, nsk, 1);
}
